// RPN_49959059587683
// MI455X (gfx1250) — compile-verified, matrix operand order checked by reading, not run on hardware
//
#include <hip/hip_runtime.h>

// ---------------------------------------------------------------------------
// RPN head for MI455X (gfx1250, wave32, WMMA + TDM).
// Dominant cost: 3x3 conv 512->512 on 100x160 = implicit GEMM
//   M=16000, N=512, K=4608  (~75.5 GFLOP)  -> v_wmma_f32_16x16x32_f16
// Memory traffic (~45 MB) is negligible at 23.3 TB/s => compute bound, so the
// matrix pipe runs f16 inputs with f32 accumulation (rel err ~5e-4 at K=4608).
// B (weights) tile is DMA'd into LDS by the Tensor Data Mover (TENSORcnt,
// s_wait_tensorcnt), double-buffered against WMMA compute; TDM hardware
// padding spreads the 128B LDS rows to 144B so b128 fragment reads are
// LDS-bank-conflict free. A stays on direct global_load_b128 (already
// software-pipelined against WMMA by the scheduler).
// ---------------------------------------------------------------------------

typedef __attribute__((ext_vector_type(16))) _Float16 v16h;
typedef __attribute__((ext_vector_type(8)))  _Float16 v8h;
typedef __attribute__((ext_vector_type(8)))  float    v8f;
typedef __attribute__((ext_vector_type(4)))  unsigned int v4u;
typedef __attribute__((ext_vector_type(8)))  int v8i;
typedef __attribute__((ext_vector_type(4)))  int v4i;

#define HF 100
#define WF 160
#define NPIX 16000           // HF*WF
#define CIN 512
#define KTOT 4608            // 9*CIN
#define NANCH 144000         // NPIX*9
#define NCH48 48             // 9 cls + 36 reg, padded to 48
#define PRE_NMS 6000
#define POST_NMS 300
#define IMH 1600.0f
#define IMW 2560.0f
#define NEGC (-1.0e9f)
#define HP 102               // padded height
#define WP 162               // padded width
#define KCG 64               // K-chunk staged to LDS per TDM stage
#define LROW (KCG + 8)       // LDS row stride in halves (144B: bank-conflict free)
#define NSTAGE 72            // 9 taps * (512/KCG)

// Use the TDM builtin when the toolchain has it (arity differs by toolchain).
#if defined(__HIP_DEVICE_COMPILE__) && \
    __has_builtin(__builtin_amdgcn_tensor_load_to_lds) && \
    __has_builtin(__builtin_amdgcn_s_wait_tensorcnt)
#define USE_TDM 1
#else
#define USE_TDM 0
#endif

// ---------------------------------------------------------------------------
// Prep: fp32 feature map -> f16 with 1-pixel zero halo.
__global__ __launch_bounds__(256) void k_pad_f16(const float* __restrict__ feat,
                                                 _Float16* __restrict__ xp,
                                                 int total) {
  int idx = blockIdx.x * 256 + threadIdx.x;
  if (idx >= total) return;
  int c   = idx & (CIN - 1);
  int rem = idx >> 9;
  int wp  = rem % WP;
  int hp  = rem / WP;
  float v = 0.0f;
  if (hp >= 1 && hp <= HF && wp >= 1 && wp <= WF)
    v = feat[(size_t)(((hp - 1) * WF) + (wp - 1)) * CIN + c];
  xp[idx] = (_Float16)v;
}

// W1 (HWIO, [k=tap*512+ci][n]) -> Wt[n][k] f16 (B operand, N-major)
__global__ __launch_bounds__(256) void k_w1_f16(const float* __restrict__ W1,
                                                _Float16* __restrict__ wt,
                                                int total) {
  int idx = blockIdx.x * 256 + threadIdx.x;
  if (idx >= total) return;
  int k = idx % KTOT;
  int n = idx / KTOT;
  wt[idx] = (_Float16)W1[(size_t)k * CIN + n];
}

// Fused cls(9) + reg(36) 1x1 weights -> Wcr[n=0..47][k=0..511] f16
__global__ __launch_bounds__(256) void k_wcr_f16(const float* __restrict__ Wc,
                                                 const float* __restrict__ Wr,
                                                 _Float16* __restrict__ wcr) {
  int idx = blockIdx.x * 256 + threadIdx.x;
  if (idx >= NCH48 * CIN) return;
  int k = idx & (CIN - 1);
  int n = idx >> 9;
  float v = 0.0f;
  if (n < 9)       v = Wc[(size_t)k * 9 + n];
  else if (n < 45) v = Wr[(size_t)k * 36 + (n - 9)];
  wcr[idx] = (_Float16)v;
}

// ---------------------------------------------------------------------------
// Fragment load matching the CDNA5 16-bit A/B layout (ISA 7.12.2):
//   lanes 0-15  : row/col = lane,    K = {base..base+7, base+16..base+23}
//   lanes 16-31 : row/col = lane-16, K = {base+8..+15,  base+24..+31}
// i.e. per lane: two contiguous 16B chunks at p and p+16 (hi lanes add +8).
__device__ __forceinline__ v16h ld_frag(const _Float16* p) {
  v8h lo = *(const v8h*)p;
  v8h hi = *(const v8h*)(p + 16);
  return __builtin_shufflevector(lo, hi, 0, 1, 2, 3, 4, 5, 6, 7,
                                 8, 9, 10, 11, 12, 13, 14, 15);
}

// ---------------------------------------------------------------------------
// Stage one B tile (128 rows x KCG halves) from global Wt into LDS.
// TDM path: one wave-level tensor_load_to_lds (16 KB DMA, TENSORcnt-tracked),
// with hardware padding of 4 DWORDs every 32 DWORDs -> LROW-half rows.
// Fallback: cooperative 16B vector copies by all 256 threads.
__device__ __forceinline__ void stage_B(const _Float16* __restrict__ gsrc,
                                        _Float16* __restrict__ sbuf, int tid) {
#if USE_TDM
  if (tid < 32) {
    unsigned long long ga = (unsigned long long)(uintptr_t)gsrc;
    unsigned lds = (unsigned)(uintptr_t)sbuf;   // flat LDS addr: low 32 bits
    v4u g0;
    g0[0] = 1u;                                   // count=1, user mode
    g0[1] = lds;                                  // lds_addr
    g0[2] = (unsigned)(ga & 0xFFFFFFFFull);       // global_addr[31:0]
    g0[3] = (unsigned)((ga >> 32) & 0x01FFFFFFu)  // global_addr[56:32]
            | 0x80000000u;                        // type=2 ("image")
    v8i g1;
    g1[0] = (int)0x07110000;   // data_size=2B, pad_en, interval=32dw, pad=4dw
    g1[1] = (int)0x12000000;   // tensor_dim0[15:0]=4608 in bits[31:16]
    g1[2] = (int)0x02000000;   // dim0 hi=0 | tensor_dim1[15:0]=512
    g1[3] = (int)0x00400000;   // dim1 hi=0 | tile_dim0=64
    g1[4] = 128;               // tile_dim1=128, tile_dim2=0
    g1[5] = KTOT;              // tensor_dim0_stride = 4608
    g1[6] = 0;                 // stride hi / dim1_stride lo
    g1[7] = 0;
    v4i g2 = (v4i)0;
    v4i g3 = (v4i)0;
#if __clang_major__ >= 23
    v8i g4 = (v8i)0;
    __builtin_amdgcn_tensor_load_to_lds(g0, g1, g2, g3, g4, 0);
#else
    __builtin_amdgcn_tensor_load_to_lds(g0, g1, g2, g3, 0);
#endif
  }
#else
#pragma unroll
  for (int q = 0; q < 4; ++q) {
    int u = tid + q * 256;        // 16B units; 128 rows * 8 units/row = 1024
    int n = u >> 3;
    int kk = (u & 7) << 3;
    *(v8h*)(sbuf + n * LROW + kk) = *(const v8h*)(gsrc + (size_t)n * KTOT + kk);
  }
#endif
}

__device__ __forceinline__ void wait_B(int tid) {
#if USE_TDM
  if (tid < 32) __builtin_amdgcn_s_wait_tensorcnt(0);
#endif
  (void)tid;
}

// ---------------------------------------------------------------------------
// 3x3 conv as implicit-im2col WMMA GEMM.
// Block = 256 threads = 8 waves in a 4(M) x 2(N) grid; each wave computes a
// 32x64 output tile (2x4 C fragments). Block tile = 128 pixels x 128 chans.
// B operand double-buffered in LDS via TDM; A direct from global.
__global__ __launch_bounds__(256) void k_conv3x3_wmma(
    const _Float16* __restrict__ xp, const _Float16* __restrict__ wt,
    const float* __restrict__ b1, _Float16* __restrict__ y) {
  __shared__ _Float16 sB[2][128 * LROW];   // 2 x 18 KB

  const int tid  = threadIdx.x;
  const int lane = tid & 31;
  const int wave = tid >> 5;
  const int wvm  = wave & 3;
  const int wvn  = wave >> 2;
  const int ncol = lane & 15;
  const int hi8  = (lane >> 4) << 3;
  const int blockM = blockIdx.x * 128;
  const int blockN = blockIdx.y * 128;

  long abase[2];
#pragma unroll
  for (int i = 0; i < 2; ++i) {
    int m = blockM + wvm * 32 + i * 16 + ncol;
    int h = m / WF, w = m - h * WF;
    abase[i] = (long)(h * WP + w) * CIN + hi8;   // padded top-left of patch
  }
  int bofs[4];
#pragma unroll
  for (int j = 0; j < 4; ++j)
    bofs[j] = (wvn * 64 + j * 16 + ncol) * LROW + hi8;

  const _Float16* gB0 = wt + (size_t)blockN * KTOT;  // this block's 128 B rows

  v8f acc[2][4] = {};

  stage_B(gB0, &sB[0][0], tid);                      // prologue: stage 0

  for (int s = 0; s < NSTAGE; ++s) {
    wait_B(tid);
    __syncthreads();                                 // sB[s&1] ready for all
    if (s + 1 < NSTAGE) {
      const int tap1 = (s + 1) >> 3, kcg1 = (s + 1) & 7;
      stage_B(gB0 + tap1 * CIN + kcg1 * KCG, &sB[(s + 1) & 1][0], tid);
    }

    const int tap = s >> 3, kcg = s & 7;
    const int dy = tap / 3, dx = tap - dy * 3;
    const long tA = (long)(dy * WP + dx) * CIN + (long)kcg * KCG;
    const _Float16* bb = &sB[s & 1][0];

#pragma unroll
    for (int k2 = 0; k2 < KCG; k2 += 32) {
      v16h af[2], bf[4];
#pragma unroll
      for (int i = 0; i < 2; ++i) af[i] = ld_frag(xp + abase[i] + tA + k2);
#pragma unroll
      for (int j = 0; j < 4; ++j) bf[j] = ld_frag(bb + bofs[j] + k2);
#pragma unroll
      for (int i = 0; i < 2; ++i)
#pragma unroll
        for (int j = 0; j < 4; ++j)
          acc[i][j] = __builtin_amdgcn_wmma_f32_16x16x32_f16(
              false, af[i], false, bf[j], (short)0, acc[i][j], false, false);
    }
    __syncthreads();                                 // protect sB[s&1] reuse
  }

  // Epilogue: bias + ReLU, store f16 activation for the 1x1 heads.
  // C layout: VGPR r -> M = r + 8*(lane>=16), N = lane&15.
#pragma unroll
  for (int j = 0; j < 4; ++j) {
    const int n = blockN + wvn * 64 + j * 16 + ncol;
    const float bias = b1[n];
#pragma unroll
    for (int i = 0; i < 2; ++i) {
      const int mb = blockM + wvm * 32 + i * 16 + hi8;
#pragma unroll
      for (int r = 0; r < 8; ++r) {
        float v = acc[i][j][r] + bias;
        v = v > 0.0f ? v : 0.0f;
        y[(long)(mb + r) * CIN + n] = (_Float16)v;
      }
    }
  }
}

// ---------------------------------------------------------------------------
// Fused 1x1 heads: GEMM M=16000, N=48 (9 cls + 36 reg + 3 pad), K=512.
// 8 waves per block, each wave owns one 16-row M tile x all 3 N tiles.
__global__ __launch_bounds__(256) void k_conv1x1_wmma(
    const _Float16* __restrict__ y, const _Float16* __restrict__ wcr,
    const float* __restrict__ bc, const float* __restrict__ br,
    float* __restrict__ logits) {
  const int lane = threadIdx.x & 31;
  const int wave = threadIdx.x >> 5;
  const int ncol = lane & 15;
  const int hi8  = (lane >> 4) << 3;
  const int mtile = blockIdx.x * 128 + wave * 16;

  const long abase = (long)(mtile + ncol) * CIN + hi8;
  long bbase[3];
#pragma unroll
  for (int j = 0; j < 3; ++j) bbase[j] = (long)(j * 16 + ncol) * CIN + hi8;

  v8f acc[3] = {};
#pragma unroll 4
  for (int kc = 0; kc < CIN; kc += 32) {
    v16h af = ld_frag(y + abase + kc);
#pragma unroll
    for (int j = 0; j < 3; ++j) {
      v16h bf = ld_frag(wcr + bbase[j] + kc);
      acc[j] = __builtin_amdgcn_wmma_f32_16x16x32_f16(
          false, af, false, bf, (short)0, acc[j], false, false);
    }
  }
#pragma unroll
  for (int j = 0; j < 3; ++j) {
    const int n = j * 16 + ncol;
    const float bias = (n < 9) ? bc[n] : ((n < 45) ? br[n - 9] : 0.0f);
#pragma unroll
    for (int r = 0; r < 8; ++r)
      logits[(long)(mtile + hi8 + r) * NCH48 + n] = acc[j][r] + bias;
  }
}

// ---------------------------------------------------------------------------
// Per-pixel softmax over the 9 cls channels + anchor box decode.
__global__ __launch_bounds__(256) void k_softmax_decode(
    const float* __restrict__ logits, const float* __restrict__ am,
    float* __restrict__ outScores, float* __restrict__ outDeltas,
    float* __restrict__ props) {
  int pix = blockIdx.x * 256 + threadIdx.x;
  if (pix >= NPIX) return;
  const float* L = logits + (long)pix * NCH48;

  float mx = L[0];
#pragma unroll
  for (int a = 1; a < 9; ++a) mx = fmaxf(mx, L[a]);
  float e[9], s = 0.0f;
#pragma unroll
  for (int a = 0; a < 9; ++a) { e[a] = expf(L[a] - mx); s += e[a]; }
  const float inv = 1.0f / s;
#pragma unroll
  for (int a = 0; a < 9; ++a) outScores[(long)pix * 9 + a] = e[a] * inv;

  const float* AMp = am + (long)pix * 36;
#pragma unroll
  for (int a = 0; a < 9; ++a) {
    const float d0 = L[9 + a * 4 + 0];
    const float d1 = L[9 + a * 4 + 1];
    const float d2 = L[9 + a * 4 + 2];
    const float d3 = L[9 + a * 4 + 3];
    outDeltas[(long)pix * 36 + a * 4 + 0] = d0;
    outDeltas[(long)pix * 36 + a * 4 + 1] = d1;
    outDeltas[(long)pix * 36 + a * 4 + 2] = d2;
    outDeltas[(long)pix * 36 + a * 4 + 3] = d3;
    const float cy = AMp[a * 4 + 0], cx = AMp[a * 4 + 1];
    const float ah = AMp[a * 4 + 2], aw = AMp[a * 4 + 3];
    const float ny = cy + d0 * ah, nx = cx + d1 * aw;
    const float sy = ah * expf(d2), sx = aw * expf(d3);
    const long pi = (long)(pix * 9 + a) * 4;
    props[pi + 0] = ny - 0.5f * sy;
    props[pi + 1] = nx - 0.5f * sx;
    props[pi + 2] = ny + 0.5f * sy;
    props[pi + 3] = nx + 0.5f * sx;
  }
}

// ---------------------------------------------------------------------------
// Top-6000 selection via binary search on the score threshold, then
// compaction with clipping and the >=16px validity filter.
__global__ __launch_bounds__(1024) void k_topk_select(
    const float* __restrict__ scores, const float* __restrict__ props,
    float* __restrict__ selS, float* __restrict__ selB, int* __restrict__ cnt) {
  __shared__ int sred[1024];
  const int tid = threadIdx.x;

  float lo = 0.0f, hi = 1.0f;
  for (int it = 0; it < 30; ++it) {
    const float mid = 0.5f * (lo + hi);
    int c = 0;
    for (int i = tid; i < NANCH; i += 1024) c += (scores[i] > mid);
    sred[tid] = c;
    __syncthreads();
    for (int s = 512; s > 0; s >>= 1) {
      if (tid < s) sred[tid] += sred[tid + s];
      __syncthreads();
    }
    const int total = sred[0];
    __syncthreads();
    if (total >= PRE_NMS) lo = mid; else hi = mid;
  }
  const float thr = lo;

  if (tid == 0) *cnt = 0;
  __syncthreads();
  for (int i = tid; i < NANCH; i += 1024) {
    const float sc = scores[i];
    if (sc > thr) {
      const int pos = atomicAdd(cnt, 1);
      if (pos < PRE_NMS) {
        const float y1 = fmaxf(props[4 * (long)i + 0], 0.0f);
        const float x1 = fmaxf(props[4 * (long)i + 1], 0.0f);
        const float y2 = fminf(props[4 * (long)i + 2], IMH);
        const float x2 = fminf(props[4 * (long)i + 3], IMW);
        const bool valid = (y2 - y1 >= 16.0f) && (x2 - x1 >= 16.0f);
        selS[pos] = valid ? sc : NEGC;
        selB[4 * pos + 0] = y1;
        selB[4 * pos + 1] = x1;
        selB[4 * pos + 2] = y2;
        selB[4 * pos + 3] = x2;
      }
    }
  }
  __syncthreads();
  int c0 = *cnt;
  if (c0 > PRE_NMS) c0 = PRE_NMS;
  for (int i = c0 + tid; i < PRE_NMS; i += 1024) {
    selS[i] = NEGC;
    selB[4 * i + 0] = 0.0f; selB[4 * i + 1] = 0.0f;
    selB[4 * i + 2] = 0.0f; selB[4 * i + 3] = 0.0f;
  }
}

// ---------------------------------------------------------------------------
// Sequential NMS, 300 iterations of block-wide argmax + suppression.
__global__ __launch_bounds__(1024) void k_nms(
    float* __restrict__ selS, const float* __restrict__ selB,
    float* __restrict__ outP) {
  __shared__ float sv[1024];
  __shared__ int si[1024];
  const int tid = threadIdx.x;

  for (int it = 0; it < POST_NMS; ++it) {
    float bv = 2.0f * NEGC;
    int bi = 0;
    for (int i = tid; i < PRE_NMS; i += 1024) {
      const float v = selS[i];
      if (v > bv) { bv = v; bi = i; }
    }
    sv[tid] = bv; si[tid] = bi;
    __syncthreads();
    for (int s = 512; s > 0; s >>= 1) {
      if (tid < s && sv[tid + s] > sv[tid]) { sv[tid] = sv[tid + s]; si[tid] = si[tid + s]; }
      __syncthreads();
    }
    const int best = si[0];
    const float bscore = sv[0];
    __syncthreads();

    const float by1 = selB[4 * best + 0], bx1 = selB[4 * best + 1];
    const float by2 = selB[4 * best + 2], bx2 = selB[4 * best + 3];
    const bool ok = bscore > NEGC * 0.5f;
    const float barea = (by2 - by1) * (bx2 - bx1);

    if (ok) {
      for (int i = tid; i < PRE_NMS; i += 1024) {
        const float y1 = selB[4 * i + 0], x1 = selB[4 * i + 1];
        const float y2 = selB[4 * i + 2], x2 = selB[4 * i + 3];
        const float iy1 = fmaxf(y1, by1), ix1 = fmaxf(x1, bx1);
        const float iy2 = fminf(y2, by2), ix2 = fminf(x2, bx2);
        const float inter = fmaxf(iy2 - iy1, 0.0f) * fmaxf(ix2 - ix1, 0.0f);
        const float area = (y2 - y1) * (x2 - x1);
        const float iou = inter / (area + barea - inter + 1e-8f);
        if (iou > 0.7f) selS[i] = NEGC;
      }
    }
    if (tid == 0) {
      selS[best] = NEGC;
      outP[4 * it + 0] = ok ? by1 : 0.0f;
      outP[4 * it + 1] = ok ? bx1 : 0.0f;
      outP[4 * it + 2] = ok ? by2 : 0.0f;
      outP[4 * it + 3] = ok ? bx2 : 0.0f;
    }
    __syncthreads();
  }
}

// ---------------------------------------------------------------------------
extern "C" void kernel_launch(void* const* d_in, const int* in_sizes, int n_in,
                              void* d_out, int out_size, void* d_ws, size_t ws_size,
                              hipStream_t stream) {
  (void)in_sizes; (void)n_in; (void)out_size; (void)ws_size;
  // setup_inputs order: image, feature_map, anchor_map, W1, b1, Wc, bc, Wr, br, training
  const float* feat = (const float*)d_in[1];
  const float* am   = (const float*)d_in[2];
  const float* W1   = (const float*)d_in[3];
  const float* b1   = (const float*)d_in[4];
  const float* Wc   = (const float*)d_in[5];
  const float* bc   = (const float*)d_in[6];
  const float* Wr   = (const float*)d_in[7];
  const float* br   = (const float*)d_in[8];
  float* out = (float*)d_out;
  char* ws = (char*)d_ws;

  size_t off = 0;
  _Float16* xp  = (_Float16*)(ws + off); off += (size_t)HP * WP * CIN * 2;   // 16.9 MB
  _Float16* wt  = (_Float16*)(ws + off); off += (size_t)CIN * KTOT * 2;      // 4.7 MB
  _Float16* y   = (_Float16*)(ws + off); off += (size_t)NPIX * CIN * 2;      // 16.4 MB
  _Float16* wcr = (_Float16*)(ws + off); off += (size_t)NCH48 * CIN * 2;     // 48 KB
  float* logits = (float*)(ws + off);    off += (size_t)NPIX * NCH48 * 4;    // 3.1 MB
  float* props  = (float*)(ws + off);    off += (size_t)NANCH * 4 * 4;       // 2.3 MB
  float* selS   = (float*)(ws + off);    off += (size_t)PRE_NMS * 4;
  float* selB   = (float*)(ws + off);    off += (size_t)PRE_NMS * 16;
  int*   cnt    = (int*)(ws + off);      off += 16;

  const int totPad = HP * WP * CIN;
  k_pad_f16<<<(totPad + 255) / 256, 256, 0, stream>>>(feat, xp, totPad);
  const int totW1 = CIN * KTOT;
  k_w1_f16<<<(totW1 + 255) / 256, 256, 0, stream>>>(W1, wt, totW1);
  k_wcr_f16<<<(NCH48 * CIN + 255) / 256, 256, 0, stream>>>(Wc, Wr, wcr);

  k_conv3x3_wmma<<<dim3(NPIX / 128, CIN / 128), 256, 0, stream>>>(xp, wt, b1, y);
  k_conv1x1_wmma<<<NPIX / 128, 256, 0, stream>>>(y, wcr, bc, br, logits);
  k_softmax_decode<<<(NPIX + 255) / 256, 256, 0, stream>>>(logits, am, out, out + NANCH, props);
  k_topk_select<<<1, 1024, 0, stream>>>(out, props, selS, selB, cnt);
  k_nms<<<1, 1024, 0, stream>>>(selS, selB, out + NANCH + (size_t)NANCH * 4);
}
